// EncoderLayer_23029614641297
// MI455X (gfx1250) — matrix operand order repaired and checked by reading, not compiled, not run on hardware
//
#include <hip/hip_runtime.h>
#include <hip/hip_bf16.h>

typedef __attribute__((ext_vector_type(16))) _Float16 v16h;
typedef __attribute__((ext_vector_type(8)))  _Float16 v8h;
typedef __attribute__((ext_vector_type(8)))  float    v8f;

#define B_   8
#define L_   1024
#define DM_  512
#define H_   8
#define DK_  64
#define DV_  512
#define MR_  (B_ * L_)          // 8192 rows (b*L + l)
#define HDK_ (H_ * DK_)         // 512
#define HDV_ (H_ * DV_)         // 4096

static __device__ __forceinline__ v16h pack16(v8h lo, v8h hi) {
  v16h r;
#pragma unroll
  for (int i = 0; i < 8; ++i) { r[i] = lo[i]; r[8 + i] = hi[i]; }
  return r;
}

// A-fragment (16x32 f16): lane holds row = lane%16; K chunks {hs*8..+7} and {16+hs*8..+7}
static __device__ __forceinline__ v16h load_afrag(const _Float16* base, int ld, int row,
                                                  int k0, int hs) {
  const _Float16* p = base + (size_t)row * ld + k0 + hs * 8;
  v8h lo = *(const v8h*)p;
  v8h hi = *(const v8h*)(p + 16);
  return pack16(lo, hi);
}

// B-fragment (32x16 f16) from row-major W (N x K): lane holds col n = lane%16 -> W row n,
// K chunks {k0+hs*8..+7} and {k0+16+hs*8..+7}
static __device__ __forceinline__ v16h load_bfrag(const _Float16* base, int ld, int n,
                                                  int k0, int hs) {
  const _Float16* p = base + (size_t)n * ld + k0 + hs * 8;
  v8h lo = *(const v8h*)p;
  v8h hi = *(const v8h*)(p + 16);
  return pack16(lo, hi);
}

static __device__ __forceinline__ v8f wmma_f16(v16h a, v16h b, v8f c) {
  return __builtin_amdgcn_wmma_f32_16x16x32_f16(false, a, false, b, (short)0, c, false, false);
}

// ---------------------------------------------------------------------------
// f32 -> f16 conversion
// ---------------------------------------------------------------------------
__global__ void k_cvt_f32_f16(const float* __restrict__ src, _Float16* __restrict__ dst, int n) {
  int i = blockIdx.x * 256 + threadIdx.x;
  if (i < n) dst[i] = (_Float16)src[i];
}

// ---------------------------------------------------------------------------
// Generic WMMA GEMM: C[z] = A[z] (MxK f16, row major) * W[z]^T (W is NxK f16) + bias[z]
// One wave computes a 32x64 tile (2 A-frags x 4 B-frags = 8 WMMA per K-step).
// Optional transposed store: C^T[n*ldT + m] (used for V).
// ---------------------------------------------------------------------------
__global__ void __launch_bounds__(32)
k_gemm_f16(const _Float16* __restrict__ A, const _Float16* __restrict__ W,
           const float* __restrict__ bias, _Float16* __restrict__ C,
           int M, int N, int K,
           long sA, long sW, long sB, long sC,
           int transposed, int ldT) {
  const int lane = threadIdx.x & 31;
  const int ln = lane & 15, hs = lane >> 4;
  const int m0 = blockIdx.x * 32;
  const int n0 = blockIdx.y * 64;
  const int z  = blockIdx.z;

  const _Float16* Az = A + (size_t)z * sA;
  const _Float16* Wz = W + (size_t)z * sW;
  const float*    bz = bias + (size_t)z * sB;
  _Float16*       Cz = C + (size_t)z * sC;

  v8f acc[2][4];
#pragma unroll
  for (int i = 0; i < 2; ++i)
#pragma unroll
    for (int j = 0; j < 4; ++j) acc[i][j] = (v8f){};

#pragma unroll 2
  for (int k0 = 0; k0 < K; k0 += 32) {
    if (k0 + 128 < K) {
      __builtin_prefetch(Az + (size_t)(m0 + ln) * K + k0 + 128, 0, 3);
      __builtin_prefetch(Az + (size_t)(m0 + 16 + ln) * K + k0 + 128, 0, 3);
    }
    v16h a0 = load_afrag(Az, K, m0 + ln, k0, hs);
    v16h a1 = load_afrag(Az, K, m0 + 16 + ln, k0, hs);
#pragma unroll
    for (int j = 0; j < 4; ++j) {
      v16h b = load_bfrag(Wz, K, n0 + j * 16 + ln, k0, hs);
      acc[0][j] = wmma_f16(a0, b, acc[0][j]);
      acc[1][j] = wmma_f16(a1, b, acc[1][j]);
    }
  }

  if (!transposed) {
#pragma unroll
    for (int i = 0; i < 2; ++i)
#pragma unroll
      for (int j = 0; j < 4; ++j) {
        int n = n0 + j * 16 + ln;
        float bn = bz[n];
#pragma unroll
        for (int r = 0; r < 8; ++r) {
          int m = m0 + i * 16 + r + hs * 8;
          Cz[(size_t)m * N + n] = (_Float16)(acc[i][j][r] + bn);
        }
      }
  } else {
    // rows m0+i*16+hs*8 .. +7 are contiguous in C^T -> one 16B store per tile
#pragma unroll
    for (int i = 0; i < 2; ++i)
#pragma unroll
      for (int j = 0; j < 4; ++j) {
        int n = n0 + j * 16 + ln;
        float bn = bz[n];
        v8h pk;
#pragma unroll
        for (int r = 0; r < 8; ++r) pk[r] = (_Float16)(acc[i][j][r] + bn);
        *(v8h*)(Cz + (size_t)n * ldT + m0 + i * 16 + hs * 8) = pk;
      }
  }
}

// ---------------------------------------------------------------------------
// Flash attention: one wave per (q-tile 32, h*b, DV-slice 128).
// Q,K: (B*L) x (H*DK) f16 row major. Vt: (H*DV) x (B*L) f16 (transposed).
// OutH: (H, B*L, DV) f16. slf_mask: (B,L,L) bool, key-position only.
// K-fragments and V-fragments are shared across both q-subtiles (2x reuse).
// ---------------------------------------------------------------------------
__global__ void __launch_bounds__(32)
k_attention(const _Float16* __restrict__ Q, const _Float16* __restrict__ Km,
            const _Float16* __restrict__ Vt, const unsigned char* __restrict__ slf,
            _Float16* __restrict__ OutH) {
  __shared__ _Float16 Pt[32 * 32];

  const int lane = threadIdx.x & 31;
  const int ln = lane & 15, hs = lane >> 4;
  const int q0 = blockIdx.x * 32;
  const int h  = blockIdx.y / B_;
  const int b  = blockIdx.y % B_;
  const int e0 = blockIdx.z * 128;

  const _Float16* Qb = Q  + (size_t)(b * L_) * HDK_ + h * DK_;
  const _Float16* Kb = Km + (size_t)(b * L_) * HDK_ + h * DK_;
  const _Float16* Vb = Vt + (size_t)(h * DV_ + e0) * MR_ + (size_t)b * L_;
  const unsigned char* mb = slf + (size_t)b * L_ * L_;   // row 0: mask depends on key only

  // Q fragments: 2 q-subtiles x 2 K-steps (d = 0..31, 32..63), reused every key chunk
  v16h aq[2][2];
#pragma unroll
  for (int i = 0; i < 2; ++i) {
    aq[i][0] = load_afrag(Qb, HDK_, q0 + i * 16 + ln, 0, hs);
    aq[i][1] = load_afrag(Qb, HDK_, q0 + i * 16 + ln, 32, hs);
  }

  v8f acc[2][8];
#pragma unroll
  for (int i = 0; i < 2; ++i)
#pragma unroll
    for (int j = 0; j < 8; ++j) acc[i][j] = (v8f){};
  float mrow[2][8], lrow[2][8];
#pragma unroll
  for (int i = 0; i < 2; ++i)
#pragma unroll
    for (int r = 0; r < 8; ++r) { mrow[i][r] = -1e30f; lrow[i][r] = 0.0f; }

  for (int c = 0; c < L_; c += 32) {
    // S tiles: 32q x 32keys, scaled by 1/sqrt(DK). K frags shared by both q-subtiles.
    v8f s[2][2];
#pragma unroll
    for (int i = 0; i < 2; ++i) { s[i][0] = (v8f){}; s[i][1] = (v8f){}; }
    {
      v16h bk00 = load_bfrag(Kb, HDK_, c + ln, 0, hs);
      v16h bk01 = load_bfrag(Kb, HDK_, c + ln, 32, hs);
      v16h bk10 = load_bfrag(Kb, HDK_, c + 16 + ln, 0, hs);
      v16h bk11 = load_bfrag(Kb, HDK_, c + 16 + ln, 32, hs);
#pragma unroll
      for (int i = 0; i < 2; ++i) {
        s[i][0] = wmma_f16(aq[i][0], bk00, s[i][0]);
        s[i][0] = wmma_f16(aq[i][1], bk01, s[i][0]);
        s[i][1] = wmma_f16(aq[i][0], bk10, s[i][1]);
        s[i][1] = wmma_f16(aq[i][1], bk11, s[i][1]);
      }
    }
    const bool pad0 = mb[c + ln] != 0;
    const bool pad1 = mb[c + 16 + ln] != 0;

    float alpha[2][8];
#pragma unroll
    for (int i = 0; i < 2; ++i)
#pragma unroll
      for (int r = 0; r < 8; ++r) {
        float x0 = pad0 ? -1e30f : s[i][0][r] * 0.125f;
        float x1 = pad1 ? -1e30f : s[i][1][r] * 0.125f;
        // row max across the 16 lanes of this half (cols 0..31 of the chunk)
        float mx = fmaxf(x0, x1);
#pragma unroll
        for (int off = 1; off < 16; off <<= 1) mx = fmaxf(mx, __shfl_xor(mx, off, 32));
        float mn = fmaxf(mrow[i][r], mx);
        float p0 = expf(x0 - mn);
        float p1 = expf(x1 - mn);
        float sum = p0 + p1;
#pragma unroll
        for (int off = 1; off < 16; off <<= 1) sum += __shfl_xor(sum, off, 32);
        alpha[i][r] = expf(mrow[i][r] - mn);
        lrow[i][r]  = lrow[i][r] * alpha[i][r] + sum;
        mrow[i][r]  = mn;
        s[i][0][r] = p0;
        s[i][1][r] = p1;
      }
#pragma unroll
    for (int i = 0; i < 2; ++i)
#pragma unroll
      for (int j = 0; j < 8; ++j)
#pragma unroll
        for (int r = 0; r < 8; ++r) acc[i][j][r] *= alpha[i][r];

    // C-layout P -> LDS (32x32) -> A-fragments. DS ops are in-order within a wave.
#pragma unroll
    for (int i = 0; i < 2; ++i)
#pragma unroll
      for (int r = 0; r < 8; ++r) {
        Pt[(i * 16 + r + hs * 8) * 32 + ln]      = (_Float16)s[i][0][r];
        Pt[(i * 16 + r + hs * 8) * 32 + 16 + ln] = (_Float16)s[i][1][r];
      }
#if __has_builtin(__builtin_amdgcn_s_wait_dscnt)
    __builtin_amdgcn_s_wait_dscnt(0);
#else
    asm volatile("s_wait_dscnt 0" ::: "memory");
#endif
    v16h ap[2];
#pragma unroll
    for (int i = 0; i < 2; ++i) {
      const _Float16* pp = Pt + (i * 16 + ln) * 32 + hs * 8;
      v8h lo = *(const v8h*)pp;
      v8h hi = *(const v8h*)(pp + 16);
      ap[i] = pack16(lo, hi);
    }
    // PV: 16 x 128 per q-subtile; each V fragment reused by both q-subtiles
#pragma unroll
    for (int j = 0; j < 8; ++j) {
      v16h bv = load_bfrag(Vb, MR_, j * 16 + ln, c, hs);
      acc[0][j] = wmma_f16(ap[0], bv, acc[0][j]);
      acc[1][j] = wmma_f16(ap[1], bv, acc[1][j]);
    }
  }

#pragma unroll
  for (int i = 0; i < 2; ++i) {
    _Float16* ob = OutH + (size_t)h * MR_ * DV_ + (size_t)(b * L_ + q0 + i * 16) * DV_ + e0;
#pragma unroll
    for (int r = 0; r < 8; ++r) {
      float inv = 1.0f / lrow[i][r];
#pragma unroll
      for (int j = 0; j < 8; ++j)
        ob[(size_t)(r + hs * 8) * DV_ + j * 16 + ln] = (_Float16)(acc[i][j][r] * inv);
    }
  }
}

// ---------------------------------------------------------------------------
// Head-gate softmax + weighted sum: Y[m,e] = sum_h softmax_h(G[h,m,e]) * OutH[h,m,e]
// ---------------------------------------------------------------------------
__global__ void k_combine(const _Float16* __restrict__ G, const _Float16* __restrict__ OutH,
                          _Float16* __restrict__ Y) {
  int idx = blockIdx.x * 256 + threadIdx.x;
  if (idx >= MR_ * DV_) return;
  const size_t st = (size_t)MR_ * DV_;
  float g[H_], mx = -1e30f;
#pragma unroll
  for (int h = 0; h < H_; ++h) { g[h] = (float)G[(size_t)h * st + idx]; mx = fmaxf(mx, g[h]); }
  float s = 0.0f;
#pragma unroll
  for (int h = 0; h < H_; ++h) { g[h] = expf(g[h] - mx); s += g[h]; }
  float inv = 1.0f / s, y = 0.0f;
#pragma unroll
  for (int h = 0; h < H_; ++h) y += g[h] * (float)OutH[(size_t)h * st + idx];
  Y[idx] = (_Float16)(y * inv);
}

// ---------------------------------------------------------------------------
// FC + bias + residual + pad-mask, f32 output. M=MR_, N=DM_, K=DV_. 32x64 tile.
// ---------------------------------------------------------------------------
__global__ void __launch_bounds__(32)
k_fc_out(const _Float16* __restrict__ A, const _Float16* __restrict__ W,
         const float* __restrict__ bias, const float* __restrict__ residual,
         const unsigned char* __restrict__ pad, float* __restrict__ Out) {
  const int lane = threadIdx.x & 31;
  const int ln = lane & 15, hs = lane >> 4;
  const int m0 = blockIdx.x * 32;
  const int n0 = blockIdx.y * 64;

  v8f acc[2][4];
#pragma unroll
  for (int i = 0; i < 2; ++i)
#pragma unroll
    for (int j = 0; j < 4; ++j) acc[i][j] = (v8f){};

#pragma unroll 2
  for (int k0 = 0; k0 < DV_; k0 += 32) {
    v16h a0 = load_afrag(A, DV_, m0 + ln, k0, hs);
    v16h a1 = load_afrag(A, DV_, m0 + 16 + ln, k0, hs);
#pragma unroll
    for (int j = 0; j < 4; ++j) {
      v16h b = load_bfrag(W, DV_, n0 + j * 16 + ln, k0, hs);
      acc[0][j] = wmma_f16(a0, b, acc[0][j]);
      acc[1][j] = wmma_f16(a1, b, acc[1][j]);
    }
  }
#pragma unroll
  for (int i = 0; i < 2; ++i)
#pragma unroll
    for (int j = 0; j < 4; ++j) {
      int n = n0 + j * 16 + ln;
      float bn = bias[n];
#pragma unroll
      for (int r = 0; r < 8; ++r) {
        int m = m0 + i * 16 + r + hs * 8;
        float v = pad[m] ? 0.0f : (acc[i][j][r] + bn + residual[(size_t)m * DM_ + n]);
        Out[(size_t)m * DM_ + n] = v;
      }
    }
}

// ---------------------------------------------------------------------------
extern "C" void kernel_launch(void* const* d_in, const int* in_sizes, int n_in,
                              void* d_out, int out_size, void* d_ws, size_t ws_size,
                              hipStream_t stream) {
  (void)in_sizes; (void)n_in; (void)out_size; (void)ws_size;
  const float* enc          = (const float*)d_in[0];
  const unsigned char* npad = (const unsigned char*)d_in[1];  // (B,L,1) bool
  const unsigned char* slf  = (const unsigned char*)d_in[2];  // (B,L,L) bool
  const float* w_q  = (const float*)d_in[3];
  const float* b_q  = (const float*)d_in[4];
  const float* w_k  = (const float*)d_in[5];
  const float* b_k  = (const float*)d_in[6];
  const float* w_v  = (const float*)d_in[7];
  const float* b_v  = (const float*)d_in[8];
  const float* w_g  = (const float*)d_in[9];
  const float* b_g  = (const float*)d_in[10];
  const float* w_fc = (const float*)d_in[11];
  const float* b_fc = (const float*)d_in[12];
  float* out = (float*)d_out;

  char* ws = (char*)d_ws;
  size_t off = 0;
  auto alloc = [&](size_t f16_elems) -> char* {
    char* p = ws + off;
    off += (f16_elems * 2 + 255) & ~(size_t)255;
    return p;
  };
  _Float16* Xh   = (_Float16*)alloc((size_t)MR_ * DM_);
  _Float16* Wqh  = (_Float16*)alloc((size_t)HDK_ * DM_);
  _Float16* Wkh  = (_Float16*)alloc((size_t)HDK_ * DM_);
  _Float16* Wvh  = (_Float16*)alloc((size_t)HDV_ * DM_);
  _Float16* Wgh  = (_Float16*)alloc((size_t)H_ * DV_ * DM_);
  _Float16* Wfh  = (_Float16*)alloc((size_t)DM_ * DV_);
  _Float16* Qh   = (_Float16*)alloc((size_t)MR_ * HDK_);
  _Float16* Kh   = (_Float16*)alloc((size_t)MR_ * HDK_);
  _Float16* VtH  = (_Float16*)alloc((size_t)HDV_ * MR_);      // transposed V
  _Float16* OutH = (_Float16*)alloc((size_t)H_ * MR_ * DV_);
  _Float16* Gh   = (_Float16*)alloc((size_t)H_ * MR_ * DV_);
  _Float16* Yh   = (_Float16*)alloc((size_t)MR_ * DV_);

  auto cvt = [&](const float* s, _Float16* d, int n) {
    k_cvt_f32_f16<<<(n + 255) / 256, 256, 0, stream>>>(s, d, n);
  };
  cvt(enc,  Xh,  MR_ * DM_);
  cvt(w_q,  Wqh, HDK_ * DM_);
  cvt(w_k,  Wkh, HDK_ * DM_);
  cvt(w_v,  Wvh, HDV_ * DM_);
  cvt(w_g,  Wgh, H_ * DV_ * DM_);
  cvt(w_fc, Wfh, DM_ * DV_);

  // Q, K projections: (8192 x 512) = X (8192x512) * W^T
  k_gemm_f16<<<dim3(MR_ / 32, HDK_ / 64, 1), 32, 0, stream>>>(
      Xh, Wqh, b_q, Qh, MR_, HDK_, DM_, 0, 0, 0, 0, 0, 0);
  k_gemm_f16<<<dim3(MR_ / 32, HDK_ / 64, 1), 32, 0, stream>>>(
      Xh, Wkh, b_k, Kh, MR_, HDK_, DM_, 0, 0, 0, 0, 0, 0);
  // V projection, stored transposed (H*DV) x (B*L)
  k_gemm_f16<<<dim3(MR_ / 32, HDV_ / 64, 1), 32, 0, stream>>>(
      Xh, Wvh, b_v, VtH, MR_, HDV_, DM_, 0, 0, 0, 0, 1, MR_);

  // Flash attention
  k_attention<<<dim3(L_ / 32, H_ * B_, DV_ / 128), 32, 0, stream>>>(Qh, Kh, VtH, slf, OutH);

  // Per-head gate GEMM, batched over h via grid.z
  k_gemm_f16<<<dim3(MR_ / 32, DV_ / 64, H_), 32, 0, stream>>>(
      OutH, Wgh, b_g, Gh, MR_, DV_, DM_,
      (long)MR_ * DV_, (long)DV_ * DM_, (long)DV_, (long)MR_ * DV_, 0, 0);

  // Gate softmax over heads + weighted sum
  k_combine<<<(MR_ * DV_ + 255) / 256, 256, 0, stream>>>(Gh, OutH, Yh);

  // FC + bias + residual + pad mask
  k_fc_out<<<dim3(MR_ / 32, DM_ / 64, 1), 32, 0, stream>>>(Yh, Wfh, b_fc, enc, npad, out);
}
